// EncoderLayer_12790412607592
// MI455X (gfx1250) — matrix operand order repaired and checked by reading, not compiled, not run on hardware
//
#include <hip/hip_runtime.h>
#include <hip/hip_bf16.h>

// Problem constants (match reference)
#define BB  8
#define LL  512
#define HH  512
#define NHH 8
#define DHH 64
#define L2  1024   // 2*L

#define LDSB_LD 40   // LDS row stride (halfs): 80B rows -> 16B aligned, bank-spread

typedef __attribute__((ext_vector_type(16))) _Float16 v16h;
typedef __attribute__((ext_vector_type(8)))  float    v8f;

// ---------------------------------------------------------------------------
// WMMA operand loaders (wave32, contiguous k = 16*half + s in both A and B)
// ---------------------------------------------------------------------------

// A matrix 16x32 f16, row-major source with row stride ld.
// lane holds row m = lane&15; half = lane>>4; VGPR v holds
// K = half*16 + 2*v + {0,1}
__device__ __forceinline__ v16h load_tile_a(const _Float16* __restrict__ p, int ld) {
  const int lane = threadIdx.x & 31;
  const int row  = lane & 15;
  const int half = lane >> 4;
  v16h out;
#pragma unroll
  for (int v = 0; v < 8; ++v) {
    const int k = (half << 4) + (v << 1);
    out[2 * v]     = p[(size_t)row * ld + k];
    out[2 * v + 1] = p[(size_t)row * ld + k + 1];
  }
  return out;
}

// B matrix 32x16 f16 where the source is stored as N x K row-major (stride ld),
// i.e. B[k][n] = p[n*ld + k]  (computes X @ P^T).
// lane holds column n = lane&15; K = (lane>>4)*16 + 2*v + {0,1}
__device__ __forceinline__ v16h load_tile_bt(const _Float16* p, int ld) {
  const int lane  = threadIdx.x & 31;
  const int col   = lane & 15;
  const int kbase = (lane >> 4) << 4;
  v16h out;
#pragma unroll
  for (int v = 0; v < 8; ++v) {
    out[2 * v]     = p[(size_t)col * ld + kbase + 2 * v];
    out[2 * v + 1] = p[(size_t)col * ld + kbase + 2 * v + 1];
  }
  return out;
}

// B matrix 32x16 f16 where the source is stored K x N row-major (stride ld).
__device__ __forceinline__ v16h load_tile_b(const _Float16* __restrict__ p, int ld) {
  const int lane  = threadIdx.x & 31;
  const int col   = lane & 15;
  const int kbase = (lane >> 4) << 4;
  v16h out;
#pragma unroll
  for (int v = 0; v < 8; ++v) {
    out[2 * v]     = p[(size_t)(kbase + 2 * v) * ld + col];
    out[2 * v + 1] = p[(size_t)(kbase + 2 * v + 1) * ld + col];
  }
  return out;
}

#define WMMA_F16(a, b, c) \
  __builtin_amdgcn_wmma_f32_16x16x32_f16(false, (a), false, (b), (short)0, (c), false, false)

// ---------------------------------------------------------------------------
// f32 -> f16 conversion
// ---------------------------------------------------------------------------
__global__ void cvt_f32_to_f16_kernel(const float* __restrict__ in,
                                      _Float16* __restrict__ out, int n) {
  int i = blockIdx.x * blockDim.x + threadIdx.x;
  if (i < n) out[i] = (_Float16)in[i];
}

// ---------------------------------------------------------------------------
// Y[M,N] = X[M,K] @ W[N,K]^T + bias   (torch Linear), f16 in, f16 out.
// Block (32,4) = 4 waves. Block computes a 128x64 output block:
//   wave ty -> rows m0 = by*128 + ty*32 (2 M-tiles), all waves share the
//   64-wide N strip n0 = bx*64 (4 N-tiles) staged in LDS once per k-step.
// 8 WMMAs per wave per k-step; W global traffic cut 4x via LDS.
// ---------------------------------------------------------------------------
__global__ void gemm_xwT_bias_kernel(const _Float16* __restrict__ X,
                                     const _Float16* __restrict__ W,
                                     const float* __restrict__ bias,
                                     _Float16* __restrict__ Y,
                                     int M, int N, int K) {
  __shared__ _Float16 lds_b[64 * LDSB_LD];
  const int n0 = blockIdx.x << 6;
  const int m0 = (blockIdx.y << 7) + (threadIdx.y << 5);
  const int t  = (threadIdx.y << 5) + threadIdx.x;        // 0..127
  const int srow = t >> 1;                                 // staging row 0..63
  const int scol = (t & 1) << 4;                           // staging col 0 / 16
  v8f acc0[4] = {};
  v8f acc1[4] = {};
  const _Float16* xp0 = X + (size_t)m0 * K;
  const _Float16* xp1 = X + (size_t)(m0 + 16) * K;
  for (int k0 = 0; k0 < K; k0 += 32) {
    __syncthreads();   // previous iteration's LDS consumers done
    {
      const _Float16* g = W + (size_t)(n0 + srow) * K + k0 + scol;
      _Float16* l = lds_b + srow * LDSB_LD + scol;
#pragma unroll
      for (int q = 0; q < 16; ++q) l[q] = g[q];
    }
    __builtin_prefetch(xp0 + k0 + 64, 0, 1);
    __builtin_prefetch(xp1 + k0 + 64, 0, 1);
    __syncthreads();   // staging visible to all waves
    v16h a0 = load_tile_a(xp0 + k0, K);
    v16h a1 = load_tile_a(xp1 + k0, K);
#pragma unroll
    for (int ns = 0; ns < 4; ++ns) {
      v16h b = load_tile_bt(lds_b + (ns << 4) * LDSB_LD, LDSB_LD);
      acc0[ns] = WMMA_F16(a0, b, acc0[ns]);
      acc1[ns] = WMMA_F16(a1, b, acc1[ns]);
    }
  }
  const int lane = threadIdx.x & 31;
  const int col  = lane & 15;
  const int rb   = (lane >> 4) << 3;
#pragma unroll
  for (int ns = 0; ns < 4; ++ns) {
    const int n = n0 + (ns << 4) + col;
    const float bv = bias[n];
#pragma unroll
    for (int v = 0; v < 8; ++v) {
      Y[(size_t)(m0 + rb + v) * N + n]      = (_Float16)(acc0[ns][v] + bv);
      Y[(size_t)(m0 + 16 + rb + v) * N + n] = (_Float16)(acc1[ns][v] + bv);
    }
  }
}

// ---------------------------------------------------------------------------
// C[b,h,j] = sum_d u[h,d] * k[b,j,h*DH+d]
// ---------------------------------------------------------------------------
__global__ void cterm_kernel(const _Float16* __restrict__ kh,
                             const float* __restrict__ u,
                             float* __restrict__ Cbuf) {
  int idx = blockIdx.x * blockDim.x + threadIdx.x;   // (b*NH+h)*L + j
  if (idx >= BB * NHH * LL) return;
  const int j = idx % LL;
  const int h = (idx / LL) % NHH;
  const int b = idx / (LL * NHH);
  const _Float16* kp = kh + ((size_t)(b * LL + j)) * HH + h * DHH;
  const float* up = u + h * DHH;
  float s = 0.f;
#pragma unroll 8
  for (int d = 0; d < DHH; ++d) s += up[d] * (float)kp[d];
  Cbuf[idx] = s;
}

// ---------------------------------------------------------------------------
// Dp[h,r] = sum_d Rp[r, h*DH+d] * v[h,d]
// ---------------------------------------------------------------------------
__global__ void dterm_kernel(const _Float16* __restrict__ rph,
                             const float* __restrict__ vvec,
                             float* __restrict__ Dp) {
  int idx = blockIdx.x * blockDim.x + threadIdx.x;   // h*1024 + r
  if (idx >= NHH * L2) return;
  const int h = idx >> 10;
  const int r = idx & (L2 - 1);
  const _Float16* rp = rph + (size_t)r * HH + h * DHH;
  const float* vp = vvec + h * DHH;
  float s = 0.f;
#pragma unroll 8
  for (int d = 0; d < DHH; ++d) s += vp[d] * (float)rp[d];
  Dp[idx] = s;
}

// ---------------------------------------------------------------------------
// Bfull[b,h,i,r] = sum_d q[b,i,h,d] * Rp[r,h,d]    (L x 2L per head), f16 out
// grid (4, 32, 64) block (32,4): wave -> 16 x 64 strip (4 r-tiles), 8 WMMAs.
// ---------------------------------------------------------------------------
__global__ void bfull_kernel(const _Float16* __restrict__ qh,
                             const _Float16* __restrict__ rph,
                             _Float16* __restrict__ bfull) {
  const int r0 = (blockIdx.x * 4 + threadIdx.y) << 6;     // 64-wide r strip
  const int i0 = blockIdx.y << 4;
  const int bh = blockIdx.z;
  const int b = bh / NHH, h = bh % NHH;
  const _Float16* qp = qh + ((size_t)(b * LL + i0)) * HH + h * DHH;
  v16h a0 = load_tile_a(qp, HH);
  v16h a1 = load_tile_a(qp + 32, HH);
  v8f acc[4] = {};
#pragma unroll
  for (int rs = 0; rs < 4; ++rs) {
    const _Float16* rp = rph + (size_t)(r0 + (rs << 4)) * HH + h * DHH;
    v16h b0 = load_tile_bt(rp, HH);
    v16h b1 = load_tile_bt(rp + 32, HH);
    acc[rs] = WMMA_F16(a0, b0, acc[rs]);
    acc[rs] = WMMA_F16(a1, b1, acc[rs]);
  }
  const int lane = threadIdx.x & 31;
  const int col  = lane & 15;
  const int rb   = (lane >> 4) << 3;
#pragma unroll
  for (int rs = 0; rs < 4; ++rs)
#pragma unroll
    for (int v = 0; v < 8; ++v)
      bfull[((size_t)bh * LL + i0 + rb + v) * L2 + r0 + (rs << 4) + col] = (_Float16)acc[rs][v];
}

// ---------------------------------------------------------------------------
// scores[b,h,i,j] = (QK^T + Bfull[..,j-i+L] + C[b,h,j] + Dp[h,j-i+L]) / 8,
// masked where j >= seq_len[b].
// grid (2, 32, 64) block (32,4): wave -> 16 x 64 strip (4 j-tiles), 8 WMMAs.
// ---------------------------------------------------------------------------
__global__ void scores_kernel(const _Float16* __restrict__ qh,
                              const _Float16* __restrict__ kh,
                              const _Float16* __restrict__ bfull,
                              const float* __restrict__ Cbuf,
                              const float* __restrict__ Dp,
                              const int* __restrict__ seq_len,
                              float* __restrict__ scores) {
  const int j0 = (blockIdx.x * 4 + threadIdx.y) << 6;     // 64-wide j strip
  const int i0 = blockIdx.y << 4;
  const int bh = blockIdx.z;
  const int b = bh / NHH, h = bh % NHH;
  const _Float16* qp = qh + ((size_t)(b * LL + i0)) * HH + h * DHH;
  v16h a0 = load_tile_a(qp, HH);
  v16h a1 = load_tile_a(qp + 32, HH);
  v8f acc[4] = {};
#pragma unroll
  for (int js = 0; js < 4; ++js) {
    const _Float16* kp = kh + ((size_t)(b * LL + j0 + (js << 4))) * HH + h * DHH;
    v16h b0 = load_tile_bt(kp, HH);
    v16h b1 = load_tile_bt(kp + 32, HH);
    acc[js] = WMMA_F16(a0, b0, acc[js]);
    acc[js] = WMMA_F16(a1, b1, acc[js]);
  }
  const int lane = threadIdx.x & 31;
  const int col  = lane & 15;
  const int rb   = (lane >> 4) << 3;
  const int sl   = seq_len[b];
#pragma unroll
  for (int js = 0; js < 4; ++js) {
    const int j = j0 + (js << 4) + col;
    const float cterm = Cbuf[(size_t)bh * LL + j];
    const bool masked = (j >= sl);
#pragma unroll
    for (int v = 0; v < 8; ++v) {
      const int i = i0 + rb + v;
      const int r = j - i + LL;                       // in [1, 1023]
      float val = acc[js][v]
                + (float)bfull[((size_t)bh * LL + i) * L2 + r]
                + cterm
                + Dp[(h << 10) + r];
      val *= 0.125f;                                  // 1/sqrt(DH)
      if (masked) val = -1e15f;
      scores[((size_t)bh * LL + i) * LL + j] = val;
    }
  }
}

// ---------------------------------------------------------------------------
// Row softmax over 512 keys; one block (128 threads) per row; f16 output.
// ---------------------------------------------------------------------------
__global__ void softmax_kernel(const float* __restrict__ scores,
                               _Float16* __restrict__ attn) {
  __shared__ float red[128];
  const size_t row = blockIdx.x;
  const int t = threadIdx.x;
  const float* p = scores + row * LL;
  float x[4];
#pragma unroll
  for (int u = 0; u < 4; ++u) x[u] = p[t + 128 * u];
  float m = fmaxf(fmaxf(x[0], x[1]), fmaxf(x[2], x[3]));
  red[t] = m; __syncthreads();
  for (int s = 64; s > 0; s >>= 1) { if (t < s) red[t] = fmaxf(red[t], red[t + s]); __syncthreads(); }
  m = red[0]; __syncthreads();
  float e[4], sum = 0.f;
#pragma unroll
  for (int u = 0; u < 4; ++u) { e[u] = __expf(x[u] - m); sum += e[u]; }
  red[t] = sum; __syncthreads();
  for (int s = 64; s > 0; s >>= 1) { if (t < s) red[t] += red[t + s]; __syncthreads(); }
  const float inv = 1.0f / red[0];
  _Float16* q = attn + row * LL;
#pragma unroll
  for (int u = 0; u < 4; ++u) q[t + 128 * u] = (_Float16)(e[u] * inv);
}

// ---------------------------------------------------------------------------
// out[b,i,h*DH+d] = sum_j attn[b,h,i,j] * v[b,j,h*DH+d]
// grid (1, 32, 64) block (32,4): d0=ty*16, i0=by*16, bh=bz;  16 WMMAs/wave
// ---------------------------------------------------------------------------
__global__ void av_kernel(const _Float16* __restrict__ attn,
                          const _Float16* __restrict__ vh,
                          float* __restrict__ outb) {
  const int d0 = threadIdx.y << 4;
  const int i0 = blockIdx.y << 4;
  const int bh = blockIdx.z;
  const int b = bh / NHH, h = bh % NHH;
  const _Float16* ap = attn + ((size_t)bh * LL + i0) * LL;
  v8f c = {};
  for (int j0 = 0; j0 < LL; j0 += 32) {
    __builtin_prefetch(ap + j0 + 64, 0, 1);
    v16h a = load_tile_a(ap + j0, LL);
    v16h bm = load_tile_b(vh + ((size_t)(b * LL + j0)) * HH + h * DHH + d0, HH);
    c = WMMA_F16(a, bm, c);
  }
  const int lane = threadIdx.x & 31;
  const int col  = lane & 15;
  const int rb   = (lane >> 4) << 3;
#pragma unroll
  for (int v = 0; v < 8; ++v)
    outb[((size_t)(b * LL + i0 + rb + v)) * HH + h * DHH + d0 + col] = c[v];
}

// ---------------------------------------------------------------------------
// res = src + attn_out; LayerNorm(eps=1e-5); exact GELU. 1 block / row.
// ---------------------------------------------------------------------------
__global__ void ln_gelu_kernel(const float* __restrict__ src,
                               const float* __restrict__ attn_out,
                               const float* __restrict__ gamma,
                               const float* __restrict__ beta,
                               float* __restrict__ dst) {
  __shared__ float red[128];
  const size_t row = blockIdx.x;
  const int t = threadIdx.x;
  const float* sp = src + row * HH;
  const float* op = attn_out + row * HH;
  float r[4];
  float s = 0.f;
#pragma unroll
  for (int u = 0; u < 4; ++u) { r[u] = sp[t + 128 * u] + op[t + 128 * u]; s += r[u]; }
  red[t] = s; __syncthreads();
  for (int k = 64; k > 0; k >>= 1) { if (t < k) red[t] += red[t + k]; __syncthreads(); }
  const float mu = red[0] * (1.0f / HH);
  __syncthreads();
  float sq = 0.f;
#pragma unroll
  for (int u = 0; u < 4; ++u) { float d = r[u] - mu; sq += d * d; }
  red[t] = sq; __syncthreads();
  for (int k = 64; k > 0; k >>= 1) { if (t < k) red[t] += red[t + k]; __syncthreads(); }
  const float inv = rsqrtf(red[0] * (1.0f / HH) + 1e-5f);
#pragma unroll
  for (int u = 0; u < 4; ++u) {
    const int c = t + 128 * u;
    float x = (r[u] - mu) * inv * gamma[c] + beta[c];
    dst[row * HH + c] = 0.5f * x * (1.0f + erff(x * 0.70710678118654752f));
  }
}

// ---------------------------------------------------------------------------
// Host-side orchestration
// ---------------------------------------------------------------------------
extern "C" void kernel_launch(void* const* d_in, const int* in_sizes, int n_in,
                              void* d_out, int out_size, void* d_ws, size_t ws_size,
                              hipStream_t stream) {
  const float* src  = (const float*)d_in[0];
  const int*   slen = (const int*)  d_in[1];
  const float* pe   = (const float*)d_in[2];
  const float* Wq   = (const float*)d_in[3];
  const float* bq   = (const float*)d_in[4];
  const float* Wk   = (const float*)d_in[5];
  const float* bk   = (const float*)d_in[6];
  const float* Wv   = (const float*)d_in[7];
  const float* bv   = (const float*)d_in[8];
  const float* Wr   = (const float*)d_in[9];
  const float* br   = (const float*)d_in[10];
  const float* u    = (const float*)d_in[11];
  const float* vv   = (const float*)d_in[12];
  const float* gam  = (const float*)d_in[13];
  const float* bet  = (const float*)d_in[14];
  float* dst = (float*)d_out;

  char* w = (char*)d_ws;
  size_t off = 0;
  auto alloc = [&](size_t bytes) -> void* {
    void* p = w + off;
    off = (off + bytes + 255) & ~(size_t)255;
    return p;
  };

  const size_t nSrc = (size_t)BB * LL * HH;     // 2,097,152
  const size_t nPe  = (size_t)L2 * HH;          // 524,288
  const size_t nW   = (size_t)HH * HH;          // 262,144

  _Float16* srch = (_Float16*)alloc(nSrc * 2);
  _Float16* peh  = (_Float16*)alloc(nPe * 2);
  _Float16* wqh  = (_Float16*)alloc(nW * 2);
  _Float16* wkh  = (_Float16*)alloc(nW * 2);
  _Float16* wvh  = (_Float16*)alloc(nW * 2);
  _Float16* wrh  = (_Float16*)alloc(nW * 2);
  _Float16* qh   = (_Float16*)alloc(nSrc * 2);
  _Float16* kh   = (_Float16*)alloc(nSrc * 2);
  _Float16* vh   = (_Float16*)alloc(nSrc * 2);
  _Float16* rph  = (_Float16*)alloc(nPe * 2);
  float*    Cbuf = (float*)alloc((size_t)BB * NHH * LL * 4);
  float*    Dp   = (float*)alloc((size_t)NHH * L2 * 4);
  _Float16* bfull= (_Float16*)alloc((size_t)BB * NHH * LL * L2 * 2);   // 64 MB
  float*    scor = (float*)alloc((size_t)BB * NHH * LL * LL * 4);      // 64 MB
  _Float16* attn = (_Float16*)alloc((size_t)BB * NHH * LL * LL * 2);   // 32 MB
  float*    outb = (float*)alloc(nSrc * 4);                            // 8 MB
  (void)ws_size; (void)n_in; (void)in_sizes; (void)out_size;

  // 1) convert f32 -> f16
  auto cvt = [&](const float* in, _Float16* out, size_t n) {
    cvt_f32_to_f16_kernel<<<dim3((unsigned)((n + 255) / 256)), dim3(256), 0, stream>>>(in, out, (int)n);
  };
  cvt(src, srch, nSrc);
  cvt(pe,  peh,  nPe);
  cvt(Wq,  wqh,  nW);
  cvt(Wk,  wkh,  nW);
  cvt(Wv,  wvh,  nW);
  cvt(Wr,  wrh,  nW);

  // 2) projections: q,k,v (4096x512x512) and Rp (1024x512x512)
  {
    dim3 blk(32, 4);
    dim3 grd(HH / 64, (BB * LL) / 128);           // (8, 32)
    gemm_xwT_bias_kernel<<<grd, blk, 0, stream>>>(srch, wqh, bq, qh, BB * LL, HH, HH);
    gemm_xwT_bias_kernel<<<grd, blk, 0, stream>>>(srch, wkh, bk, kh, BB * LL, HH, HH);
    gemm_xwT_bias_kernel<<<grd, blk, 0, stream>>>(srch, wvh, bv, vh, BB * LL, HH, HH);
    dim3 grdR(HH / 64, L2 / 128);                 // (8, 8)
    gemm_xwT_bias_kernel<<<grdR, blk, 0, stream>>>(peh, wrh, br, rph, L2, HH, HH);
  }

  // 3) C and D bias terms
  cterm_kernel<<<dim3((BB * NHH * LL) / 256), dim3(256), 0, stream>>>(kh, u, Cbuf);
  dterm_kernel<<<dim3((NHH * L2) / 256), dim3(256), 0, stream>>>(rph, vv, Dp);

  // 4) Bfull = Q_h @ Rp_h^T per (b,h)
  bfull_kernel<<<dim3(L2 / 256, LL / 16, BB * NHH), dim3(32, 4), 0, stream>>>(qh, rph, bfull);

  // 5) scores = (A + B + C + D)/8, masked
  scores_kernel<<<dim3(LL / 256, LL / 16, BB * NHH), dim3(32, 4), 0, stream>>>(
      qh, kh, bfull, Cbuf, Dp, slen, scor);

  // 6) softmax over keys
  softmax_kernel<<<dim3(BB * NHH * LL), dim3(128), 0, stream>>>(scor, attn);

  // 7) out = attn @ V per (b,h)
  av_kernel<<<dim3(1, LL / 16, BB * NHH), dim3(32, 4), 0, stream>>>(attn, vh, outb);

  // 8) residual + LayerNorm + exact GELU
  ln_gelu_kernel<<<dim3(BB * LL), dim3(128), 0, stream>>>(src, outb, gam, bet, dst);
}
